// SCNN_46291157516817
// MI455X (gfx1250) — matrix operand order repaired and checked by reading, not compiled, not run on hardware
//
#include <hip/hip_runtime.h>

// ---------------------------------------------------------------------------
// SCNN directional recurrences (down / up / right / left) on
// x:[32,48,64,128] f32; each step: next = input + conv1d_C->C_K5(prev).
//
// MI455X reasoning:
//  * ~24 GFLOP total, ~50 MB live state (fits 192 MB L2) -> bound by the
//    380-step sequential recurrence chain. One workgroup per batch (32 WGs),
//    state resident in LDS, ONE block barrier per step, zero global sync.
//  * Conv-as-shift-GEMM: state stored TRANSPOSED in LDS (prevT[l+2][c],
//    c padded 48->64); D = sum_{k=0..4} W_k[48x64] * prevT[n+k][.]; every
//    WMMA B fragment is one contiguous 32B LDS read per lane.
//  * Double-buffered LDS state (ping-pong) -> a single __syncthreads per
//    step covers both RAW (write->read) and WAR (read->write) hazards.
//  * 6 independent accumulators (3 M-tiles x 2 ci-tiles) round-robin ->
//    dependent-WMMA distance 6 >= the bf16 WMMA->WMMA 5-slot hazard.
//  * __builtin_amdgcn_sched_barrier(0) pins each A-fragment load group
//    above the consuming WMMA group so the scheduler cannot serialize
//    load->wait(0)->wmma; loads of group k+1 stay outstanding during group
//    k's WMMAs (partial dscnt waits).
//
// Scratch: 2 * 32*64*48*128 floats (~100.7 MB) in d_ws (buf0 = D then R;
// buf1 = U).
// ---------------------------------------------------------------------------

typedef __attribute__((ext_vector_type(16))) __bf16 v16bf;
typedef __attribute__((ext_vector_type(8)))  float  v8f;

namespace {

constexpr int BB = 32, CC = 48, HH = 64, WW = 128;
constexpr int KW = 5;
constexpr int CP = 64;          // channel dim padded to 2 k-tiles of 32
constexpr int NTHREADS = 256;   // 8 wave32 waves

__device__ __forceinline__ int frag_off(int k, int mt, int kt, int lane) {
  return (((k * 3 + mt) * 2 + kt) * 32 + lane) << 4;
}

// Swizzled A fragments: one contiguous 32B read per lane per fragment
// (contiguous k order within a k-tile: K = 16*half + j, same as B).
__device__ __forceinline__ void build_weights(const float* __restrict__ w,
                                              const float* __restrict__ bias,
                                              __bf16* sA, float* sBias, int tid)
{
  for (int idx = tid; idx < KW * 3 * 2 * 32 * 16; idx += NTHREADS) {
    const int j    = idx & 15;
    const int lane = (idx >> 4) & 31;
    const int kt   = (idx >> 9) & 1;
    const int km   = idx >> 10;          // k*3 + mt
    const int k    = km / 3;
    const int mt   = km - 3 * k;
    const int m    = mt * 16 + (lane & 15);
    // lanes 0-15: K(ci) = {0..15}; lanes 16-31: {16..31}
    const int ci   = kt * 32 + ((lane & 16) ? 16 : 0) + j;
    float v = 0.0f;
    if (ci < CC) v = w[(m * CC + ci) * KW + k];   // w[c_out][c_in][k]
    sA[idx] = (__bf16)v;
  }
  for (int c = tid; c < CC; c += NTHREADS) sBias[c] = bias[c];
}

// Initial recurrence slice -> transposed LDS state buffer 0 (bf16, 2-row
// zero borders, channels 48..63 zero); buffer 1 fully zeroed (its borders
// are never touched by the per-step epilogue). Mirrors slice to the pass
// output via `store`.
template <int L, class GetIn, class Store>
__device__ __forceinline__ void init_state(__bf16* sP0, __bf16* sP1, int tid,
                                           GetIn getIn, Store store)
{
  for (int i = tid; i < (L + 4) * CP; i += NTHREADS) {
    const int r = i >> 6;     // row = l + 2
    const int c = i & 63;
    const int l = r - 2;
    float v = 0.0f;
    if (c < CC && l >= 0 && l < L) { v = getIn(c, l); store(c, l, v); }
    sP0[i] = (__bf16)v;
    sP1[i] = (__bf16)0.0f;
  }
  __syncthreads();
}

// One recurrence step: out[m][n] = in(m,n) + bias[m]
//                                + sum_{ci,k} w[m][ci][k] * prev[ci][n+k-2].
// Reads sCur, writes sNxt -> one barrier per step suffices.
template <int L, class GetIn, class Store>
__device__ __forceinline__ void conv_step(const __bf16* sA, const __bf16* sCur,
                                          __bf16* sNxt, const float* sBias,
                                          int tid, GetIn getIn, Store store)
{
  constexpr int NT = L / 16;
  const int wave = tid >> 5;
  const int lane = tid & 31;
  const int half = (lane >> 4) & 1;
  const int l16  = lane & 15;
  const bool active = wave < NT;       // wave owns one N-tile, all 3 M-tiles
  const int n = wave * 16 + l16;

  if (active) {
    // Input slice -> registers (global loads overlap everything below).
    float inv[3][8];
    #pragma unroll
    for (int mt = 0; mt < 3; ++mt)
      #pragma unroll
      for (int r = 0; r < 8; ++r)
        inv[mt][r] = getIn(mt * 16 + r + half * 8, n);

    // B fragments: 16 contiguous channels at sCur[n+k][..], batched loads.
    v16bf bfrag[KW][2];
    #pragma unroll
    for (int k = 0; k < KW; ++k)
      #pragma unroll
      for (int kt = 0; kt < 2; ++kt)
        bfrag[k][kt] =
            *(const v16bf*)(sCur + (n + k) * CP + kt * 32 + half * 16);

    const v8f z = {0.f, 0.f, 0.f, 0.f, 0.f, 0.f, 0.f, 0.f};
    v8f acc[3][2];
    #pragma unroll
    for (int mt = 0; mt < 3; ++mt) { acc[mt][0] = z; acc[mt][1] = z; }

    // Software-pipelined A staging: group k+1's loads are pinned above
    // group k's WMMAs by a scheduling barrier -> distinct registers,
    // partial dscnt waits, LDS latency overlapped with the matrix pipe.
    v16bf a[2][6];
    #pragma unroll
    for (int kt = 0; kt < 2; ++kt)
      #pragma unroll
      for (int mt = 0; mt < 3; ++mt)
        a[0][kt * 3 + mt] = *(const v16bf*)(sA + frag_off(0, mt, kt, lane));

    #pragma unroll
    for (int k = 0; k < KW; ++k) {
      const int cur = k & 1, nxt = cur ^ 1;
      if (k + 1 < KW) {
        #pragma unroll
        for (int kt = 0; kt < 2; ++kt)
          #pragma unroll
          for (int mt = 0; mt < 3; ++mt)
            a[nxt][kt * 3 + mt] =
                *(const v16bf*)(sA + frag_off(k + 1, mt, kt, lane));
      }
      __builtin_amdgcn_sched_barrier(0);   // keep the loads above the WMMAs
      #pragma unroll
      for (int kt = 0; kt < 2; ++kt)
        #pragma unroll
        for (int mt = 0; mt < 3; ++mt)
          acc[mt][kt] = __builtin_amdgcn_wmma_f32_16x16x32_bf16(
              false, a[cur][kt * 3 + mt], false, bfrag[k][kt], (short)0,
              acc[mt][kt], false, false);
    }

    // Epilogue per ISA f32 C/D layout: VGPR r -> M=r(+8 upper lanes), N=l16.
    #pragma unroll
    for (int mt = 0; mt < 3; ++mt)
      #pragma unroll
      for (int r = 0; r < 8; ++r) {
        const int m = mt * 16 + r + half * 8;
        const float v = acc[mt][0][r] + acc[mt][1][r] + inv[mt][r] + sBias[m];
        store(m, n, v);
        sNxt[(n + 2) * CP + m] = (__bf16)v;   // transposed state feedback
      }
  }
  __syncthreads();   // one barrier: RAW (sNxt) and WAR (sCur) both covered
}

__global__ __launch_bounds__(NTHREADS)
void scnn_kernel(const float* __restrict__ x,
                 const float* __restrict__ down_w, const float* __restrict__ down_b,
                 const float* __restrict__ right_w, const float* __restrict__ right_b,
                 const float* __restrict__ left_w, const float* __restrict__ left_b,
                 float* __restrict__ out,
                 float* __restrict__ buf0, float* __restrict__ buf1)
{
  __shared__ __bf16 sA[KW * 3 * 2 * 32 * 16];     // 30720 B
  __shared__ __bf16 sPrevT0[(WW + 4) * CP];       // 16896 B
  __shared__ __bf16 sPrevT1[(WW + 4) * CP];       // 16896 B
  __shared__ float  sBias[CC];

  const int tid = threadIdx.x;
  const int b   = blockIdx.x;          // one workgroup per batch

  const float* xb = x    + (size_t)b * CC * HH * WW;   // [c][h][w]
  float* Db       = buf0 + (size_t)b * HH * CC * WW;   // [h][c][w]
  float* Ub       = buf1 + (size_t)b * HH * CC * WW;   // [h][c][w]
  float* Rb       = buf0 + (size_t)b * WW * CC * HH;   // [w][c][h], reuses D
  float* ob       = out  + (size_t)b * CC * HH * WW;   // [c][h][w]

  // ---------------- down: d[h] = x[h] + conv(d[h-1]) --------------------
  build_weights(down_w, down_b, sA, sBias, tid);
  __syncthreads();
  init_state<WW>(sPrevT0, sPrevT1, tid,
      [&](int c, int l) { return xb[(c * HH + 0) * WW + l]; },
      [&](int c, int l, float v) { Db[(0 * CC + c) * WW + l] = v; });
  int p = 0;
  for (int h = 1; h < HH; ++h, p ^= 1) {
    __bf16* cur = p ? sPrevT1 : sPrevT0;
    __bf16* nxt = p ? sPrevT0 : sPrevT1;
    conv_step<WW>(sA, cur, nxt, sBias, tid,
      [&](int c, int l) { return xb[(c * HH + h) * WW + l]; },
      [&](int c, int l, float v) { Db[(h * CC + c) * WW + l] = v; });
  }

  // ------- up (source quirk: reuses down_w; seq = D[61..0], D[63]) -------
  __threadfence_block();               // D stores visible across the WGP
  init_state<WW>(sPrevT0, sPrevT1, tid,
      [&](int c, int l) { return Db[((HH - 1) * CC + c) * WW + l]; },
      [&](int c, int l, float v) { Ub[((HH - 1) * CC + c) * WW + l] = v; });
  p = 0;
  for (int k = 1; k < HH; ++k, p ^= 1) {
    const int ih = (k <= HH - 2) ? (HH - 2 - k) : (HH - 1);
    const int oh = HH - 1 - k;                     // U[i] = u[H-1-i]
    __bf16* cur = p ? sPrevT1 : sPrevT0;
    __bf16* nxt = p ? sPrevT0 : sPrevT1;
    conv_step<WW>(sA, cur, nxt, sBias, tid,
      [&](int c, int l) { return Db[(ih * CC + c) * WW + l]; },
      [&](int c, int l, float v) { Ub[(oh * CC + c) * WW + l] = v; });
  }

  // ---------------- right over width ------------------------------------
  __threadfence_block();               // U stores visible across the WGP
  __syncthreads();
  build_weights(right_w, right_b, sA, sBias, tid);
  __syncthreads();
  init_state<HH>(sPrevT0, sPrevT1, tid,
      [&](int c, int h) { return Ub[(h * CC + c) * WW + 0]; },
      [&](int c, int h, float v) { Rb[(0 * CC + c) * HH + h] = v; });
  p = 0;
  for (int t = 1; t < WW; ++t, p ^= 1) {
    __bf16* cur = p ? sPrevT1 : sPrevT0;
    __bf16* nxt = p ? sPrevT0 : sPrevT1;
    conv_step<HH>(sA, cur, nxt, sBias, tid,
      [&](int c, int h) { return Ub[(h * CC + c) * WW + t]; },
      [&](int c, int h, float v) { Rb[(t * CC + c) * HH + h] = v; });
  }

  // ---------------- left (reverse over width), writes output ------------
  __threadfence_block();               // R stores visible across the WGP
  __syncthreads();
  build_weights(left_w, left_b, sA, sBias, tid);
  __syncthreads();
  init_state<HH>(sPrevT0, sPrevT1, tid,
      [&](int c, int h) { return Rb[((WW - 1) * CC + c) * HH + h]; },
      [&](int c, int h, float v) { ob[(c * HH + h) * WW + (WW - 1)] = v; });
  p = 0;
  for (int k = 1; k < WW; ++k, p ^= 1) {
    const int w = WW - 1 - k;   // input R[w]; output column w (out[.,w]=l[W-1-w])
    __bf16* cur = p ? sPrevT1 : sPrevT0;
    __bf16* nxt = p ? sPrevT0 : sPrevT1;
    conv_step<HH>(sA, cur, nxt, sBias, tid,
      [&](int c, int h) { return Rb[(w * CC + c) * HH + h]; },
      [&](int c, int h, float v) { ob[(c * HH + h) * WW + w] = v; });
  }
}

} // anonymous namespace

extern "C" void kernel_launch(void* const* d_in, const int* in_sizes, int n_in,
                              void* d_out, int out_size, void* d_ws, size_t ws_size,
                              hipStream_t stream) {
  (void)in_sizes; (void)n_in; (void)out_size; (void)ws_size;
  const float* x       = (const float*)d_in[0];
  const float* down_w  = (const float*)d_in[1];
  const float* down_b  = (const float*)d_in[2];
  const float* right_w = (const float*)d_in[3];
  const float* right_b = (const float*)d_in[4];
  const float* left_w  = (const float*)d_in[5];
  const float* left_b  = (const float*)d_in[6];
  float* out  = (float*)d_out;
  float* buf0 = (float*)d_ws;                            // D, then R
  float* buf1 = buf0 + (size_t)BB * CC * HH * WW;        // U

  scnn_kernel<<<dim3(BB), dim3(NTHREADS), 0, stream>>>(
      x, down_w, down_b, right_w, right_b, left_w, left_b, out, buf0, buf1);
}
